// CausalSelfAttention_75041668596139
// MI455X (gfx1250) — matrix operand order repaired and checked by reading, not compiled, not run on hardware
//
#include <hip/hip_runtime.h>
#include <math.h>

// ---------------- types ----------------
typedef __bf16 bf16;
typedef __attribute__((ext_vector_type(16))) __bf16       v16bf;
typedef __attribute__((ext_vector_type(8)))  float        v8f;
typedef __attribute__((ext_vector_type(4)))  unsigned int vu4;
typedef int v4i __attribute__((vector_size(16)));   // matches async-LDS builtin param

#define AS1 __attribute__((address_space(1)))
#define AS3 __attribute__((address_space(3)))

union FragU { v16bf v; vu4 q[2]; };

constexpr int B_ = 4, T_ = 2048, C_ = 1024, H_ = 16, D_ = 64;
constexpr int BT_ = B_ * T_; // 8192

// ---- async global->LDS path (gfx1250 ASYNCcnt), guarded by __has_builtin ----
#if __has_builtin(__builtin_amdgcn_global_load_async_to_lds_b128)
#define USE_ASYNC_LDS 1
#else
#define USE_ASYNC_LDS 0
#endif

#if __has_builtin(__builtin_amdgcn_s_wait_asynccnt)
#define WAIT_ASYNC(n) __builtin_amdgcn_s_wait_asynccnt(n)
#else
#define WAIT_ASYNC(n) asm volatile("s_wait_asynccnt %0" ::"i"(n) : "memory")
#endif

__device__ inline v8f vzero8() {
  v8f z = {0.f,0.f,0.f,0.f,0.f,0.f,0.f,0.f};
  return z;
}

__device__ inline v8f wmma_bf16(v16bf a, v16bf b, v8f c) {
  // V_WMMA_F32_16X16X32_BF16: D = A(16x32) * B(32x16) + C(16x16 f32)
  return __builtin_amdgcn_wmma_f32_16x16x32_bf16(
      /*neg_a=*/false, a, /*neg_b=*/false, b,
      /*c_mod=*/(short)0, c, /*reuse_a=*/false, /*reuse_b=*/false);
}

// A-fragment (16x32, bf16) from row-major [rows, stride] memory.
// ISA layout: lanes 0-15 row M=lane; elem i<8 -> K=8h+i, i>=8 -> K=16+8h+(i-8)
__device__ inline v16bf load_frag_a(const bf16* base, int stride, int row0, int kbase) {
  int lane = threadIdx.x & 31;
  int m = lane & 15, h = lane >> 4;
  const bf16* p = base + (size_t)(row0 + m) * stride + kbase + 8 * h;
  FragU f;
  f.q[0] = *(const vu4*)p;          // K = kbase + 8h + [0..7]
  f.q[1] = *(const vu4*)(p + 16);   // K = kbase + 16 + 8h + [0..7]
  return f.v;
}

// B-fragment (32x16, bf16) from B^T stored row-major [cols(N), stride(K)].
// ISA layout: lane = column N; elem i<8 -> K=8h+i, i>=8 -> K=16+8h+(i-8)
__device__ inline v16bf load_frag_b(const bf16* base, int stride, int col0, int kbase) {
  int lane = threadIdx.x & 31;
  int n = lane & 15, h = lane >> 4;
  const bf16* p = base + (size_t)(col0 + n) * stride + kbase + 8 * h;
  FragU f;
  f.q[0] = *(const vu4*)p;          // K = kbase + 8h + [0..7]
  f.q[1] = *(const vu4*)(p + 16);   // K = kbase + 16 + 8h + [0..7]
  return f.v;
}

// ---------------- conversion kernels ----------------
__global__ void f32_to_bf16_kernel(const float* __restrict__ in, bf16* __restrict__ out, int n) {
  int i = blockIdx.x * blockDim.x + threadIdx.x;
  if (i < n) out[i] = (bf16)in[i];
}

// W [K,N] f32 -> Wt [N,K] bf16
__global__ void transpose_to_bf16_kernel(const float* __restrict__ W, bf16* __restrict__ Wt,
                                         int K, int N) {
  int i = blockIdx.x * blockDim.x + threadIdx.x;
  if (i < N * K) {
    int nn = i / K, kk = i - nn * K;
    Wt[i] = (bf16)W[(size_t)kk * N + nn];
  }
}

// ---------------- bf16 GEMM: C[M,N] f32 = A[M,K] @ Bt[N,K]^T ----------------
// block tile 128x128, K-step 32, 8 waves each computing a 64x32 wave tile.
// Double-buffered LDS fed by GLOBAL_LOAD_ASYNC_TO_LDS_B128 when available.
__global__ __launch_bounds__(256) void gemm_bf16_kernel(
    const bf16* __restrict__ A, const bf16* __restrict__ Bt, float* __restrict__ C,
    int M, int N, int K) {
  const int bm = blockIdx.y * 128, bn = blockIdx.x * 128;
  const int tid = threadIdx.x;
  const int lrow = tid >> 1;
  const int lhalf = (tid & 1) * 16;
  const int w = tid >> 5;
  const int wm = (w >> 2) * 64;   // 0 or 64
  const int wn = (w & 3) * 32;    // 0,32,64,96

  v8f acc[4][2];
#pragma unroll
  for (int mi = 0; mi < 4; ++mi)
#pragma unroll
    for (int ni = 0; ni < 2; ++ni) acc[mi][ni] = vzero8();

#if USE_ASYNC_LDS
  __shared__ bf16 sA[2][128 * 32];
  __shared__ bf16 sB[2][128 * 32];

  auto issue_tile = [&](int buf, int kt) {
    const bf16* ga = A + (size_t)(bm + lrow) * K + kt + lhalf;
    const bf16* gb = Bt + (size_t)(bn + lrow) * K + kt + lhalf;
    bf16* la = &sA[buf][lrow * 32 + lhalf];
    bf16* lb = &sB[buf][lrow * 32 + lhalf];
    __builtin_amdgcn_global_load_async_to_lds_b128((AS1 v4i*)ga,       (AS3 v4i*)la,       0, 0);
    __builtin_amdgcn_global_load_async_to_lds_b128((AS1 v4i*)(ga + 8), (AS3 v4i*)(la + 8), 0, 0);
    __builtin_amdgcn_global_load_async_to_lds_b128((AS1 v4i*)gb,       (AS3 v4i*)lb,       0, 0);
    __builtin_amdgcn_global_load_async_to_lds_b128((AS1 v4i*)(gb + 8), (AS3 v4i*)(lb + 8), 0, 0);
  };

  const int ntiles = K >> 5;
  issue_tile(0, 0);
  if (ntiles > 1) issue_tile(1, 32);

  for (int i = 0; i < ntiles; ++i) {
    // 4 async ops per tile, completed in order: <=4 outstanding => tile i landed
    if (i + 1 < ntiles) WAIT_ASYNC(4); else WAIT_ASYNC(0);
    __syncthreads();

    const bf16* cA = sA[i & 1];
    const bf16* cB = sB[i & 1];
    v16bf bfr[2];
#pragma unroll
    for (int ni = 0; ni < 2; ++ni) bfr[ni] = load_frag_b(cB, 32, wn + 16 * ni, 0);
#pragma unroll
    for (int mi = 0; mi < 4; ++mi) {
      v16bf afr = load_frag_a(cA, 32, wm + 16 * mi, 0);
      acc[mi][0] = wmma_bf16(afr, bfr[0], acc[mi][0]);
      acc[mi][1] = wmma_bf16(afr, bfr[1], acc[mi][1]);
    }
    __syncthreads();                       // all waves done reading buf(i&1)
    if (i + 2 < ntiles) issue_tile(i & 1, (i + 2) * 32);
  }
#else
  __shared__ bf16 sA[128 * 32];
  __shared__ bf16 sB[128 * 32];
  for (int kt = 0; kt < K; kt += 32) {
    const bf16* ga = A + (size_t)(bm + lrow) * K + kt + lhalf;
    const bf16* gb = Bt + (size_t)(bn + lrow) * K + kt + lhalf;
    *(vu4*)&sA[lrow * 32 + lhalf]     = *(const vu4*)ga;
    *(vu4*)&sA[lrow * 32 + lhalf + 8] = *(const vu4*)(ga + 8);
    *(vu4*)&sB[lrow * 32 + lhalf]     = *(const vu4*)gb;
    *(vu4*)&sB[lrow * 32 + lhalf + 8] = *(const vu4*)(gb + 8);
    if (kt + 32 < K) {
      __builtin_prefetch(ga + 32, 0, 1);
      __builtin_prefetch(gb + 32, 0, 1);
    }
    __syncthreads();
    v16bf bfr[2];
#pragma unroll
    for (int ni = 0; ni < 2; ++ni) bfr[ni] = load_frag_b(sB, 32, wn + 16 * ni, 0);
#pragma unroll
    for (int mi = 0; mi < 4; ++mi) {
      v16bf afr = load_frag_a(sA, 32, wm + 16 * mi, 0);
      acc[mi][0] = wmma_bf16(afr, bfr[0], acc[mi][0]);
      acc[mi][1] = wmma_bf16(afr, bfr[1], acc[mi][1]);
    }
    __syncthreads();
  }
#endif

  const int lane = tid & 31, n = lane & 15, h = lane >> 4;
#pragma unroll
  for (int mi = 0; mi < 4; ++mi)
#pragma unroll
    for (int ni = 0; ni < 2; ++ni)
#pragma unroll
      for (int r = 0; r < 8; ++r) {
        int gm = bm + wm + 16 * mi + r + 8 * h;   // C layout: M = r + 8h
        int gn = bn + wn + 16 * ni + n;           // N = lane&15
        C[(size_t)gm * N + gn] = acc[mi][ni][r];
      }
}

// ---------------- RoPE + split: qkv f32 -> Qh/Kh [B,H,T,D] bf16, Vt [B,H,D,T] bf16 ----
__global__ void rope_split_kernel(const float* __restrict__ qkv,
                                  bf16* __restrict__ Qh, bf16* __restrict__ Kh,
                                  bf16* __restrict__ Vt) {
  int i = blockIdx.x * blockDim.x + threadIdx.x;   // over B*T*C
  if (i >= BT_ * C_) return;
  int d = i & 63;
  int h = (i >> 6) & 15;
  int t = (i >> 10) & (T_ - 1);
  int b = i >> 21;
  size_t row = (size_t)(b * T_ + t) * (3 * C_);
  int col = h * 64 + d;
  float q  = qkv[row + col];
  float k  = qkv[row + C_ + col];
  float v  = qkv[row + 2 * C_ + col];
  int d2 = d ^ 32;
  float q2 = qkv[row + h * 64 + d2];
  float k2 = qkv[row + C_ + h * 64 + d2];
  // inv_freq = 10000^{-(d%32)/32}; ln(10000)/32 = 0.28782313662425572
  float ang = (float)t * expf(-(float)(d & 31) * 0.28782313662425572f);
  float cs = cosf(ang), sn = sinf(ang);
  float rq = (d < 32) ? (q * cs - q2 * sn) : (q * cs + q2 * sn);
  float rk = (d < 32) ? (k * cs - k2 * sn) : (k * cs + k2 * sn);
  size_t o = ((size_t)(b * H_ + h) * T_ + t) * D_ + d;
  Qh[o] = (bf16)(rq * 0.125f);   // fold 1/sqrt(64) into Q
  Kh[o] = (bf16)rk;
  Vt[((size_t)(b * H_ + h) * D_ + d) * T_ + t] = (bf16)v;
}

// ---------------- flash attention: one wave per 16-query tile --------------
__global__ __launch_bounds__(256) void flash_attn_kernel(
    const bf16* __restrict__ Qh, const bf16* __restrict__ Kh,
    const bf16* __restrict__ Vt, bf16* __restrict__ Oh) {
  __shared__ bf16 sP[8 * 16 * 32];                 // per-wave 16x32 P tile
  const int w = threadIdx.x >> 5;
  const int lane = threadIdx.x & 31;
  const int n = lane & 15, h = lane >> 4;
  bf16* sPw = sP + w * 512;

  int wave = blockIdx.x * 8 + w;
  int qt = wave & (T_ / 16 - 1);
  int bh = wave >> 7;
  int b = bh >> 4, hh = bh & 15;
  int q0 = qt * 16;

  const bf16* Qb = Qh + ((size_t)(b * H_ + hh) * T_) * D_;
  const bf16* Kb = Kh + ((size_t)(b * H_ + hh) * T_) * D_;
  const bf16* Vb = Vt + ((size_t)(b * H_ + hh) * D_) * T_;

  v16bf aq0 = load_frag_a(Qb, D_, q0, 0);    // d 0..31
  v16bf aq1 = load_frag_a(Qb, D_, q0, 32);   // d 32..63

  float rmax[8], rsum[8];
#pragma unroll
  for (int r = 0; r < 8; ++r) { rmax[r] = -__builtin_inff(); rsum[r] = 0.f; }
  v8f co[4];
#pragma unroll
  for (int dt = 0; dt < 4; ++dt) co[dt] = vzero8();

  const int kend = q0 + 16;
  for (int j0 = 0; j0 < kend; j0 += 32) {
    // S tiles (16q x 16k each): S = Q @ K^T, A=Q, B^T = K[key][d] row-major
    v8f s0 = vzero8(), s1 = vzero8();
    {
      v16bf bk;
      bk = load_frag_b(Kb, D_, j0, 0);       s0 = wmma_bf16(aq0, bk, s0);
      bk = load_frag_b(Kb, D_, j0, 32);      s0 = wmma_bf16(aq1, bk, s0);
      bk = load_frag_b(Kb, D_, j0 + 16, 0);  s1 = wmma_bf16(aq0, bk, s1);
      bk = load_frag_b(Kb, D_, j0 + 16, 32); s1 = wmma_bf16(aq1, bk, s1);
    }
    // online softmax: row M = r + 8h, column key = j0 + {0,16} + n
#pragma unroll
    for (int r = 0; r < 8; ++r) {
      int q = q0 + r + 8 * h;
      float v0 = (j0 + n <= q)      ? s0[r] : -__builtin_inff();
      float v1 = (j0 + 16 + n <= q) ? s1[r] : -__builtin_inff();
      float bmax = fmaxf(v0, v1);
#pragma unroll
      for (int ms = 1; ms < 16; ms <<= 1) bmax = fmaxf(bmax, __shfl_xor(bmax, ms, 32));
      float mnew = fmaxf(rmax[r], bmax);
      float p0 = __expf(v0 - mnew);
      float p1 = __expf(v1 - mnew);
      float scale = __expf(rmax[r] - mnew);
      float ps = p0 + p1;
#pragma unroll
      for (int ms = 1; ms < 16; ms <<= 1) ps += __shfl_xor(ps, ms, 32);
      rsum[r] = rsum[r] * scale + ps;
      rmax[r] = mnew;
#pragma unroll
      for (int dt = 0; dt < 4; ++dt) co[dt][r] = co[dt][r] * scale;
      int prow = (r + 8 * h) * 32;
      sPw[prow + n]      = (bf16)p0;
      sPw[prow + 16 + n] = (bf16)p1;
    }
    // O += P(16x32) @ V(32x16 per d-tile); B^T = V^T[d][k] = Vt row-major
    v16bf ap = load_frag_a(sPw, 32, 0, 0);
#pragma unroll
    for (int dt = 0; dt < 4; ++dt) {
      v16bf bv = load_frag_b(Vb, T_, dt * 16, j0);
      co[dt] = wmma_bf16(ap, bv, co[dt]);
    }
  }

  // normalize and write O as bf16 [B*T, C] at (b,q, hh*64 + d)
#pragma unroll
  for (int dt = 0; dt < 4; ++dt)
#pragma unroll
    for (int r = 0; r < 8; ++r) {
      int q = q0 + r + 8 * h;
      int d = dt * 16 + n;
      Oh[(size_t)(b * T_ + q) * C_ + hh * D_ + d] = (bf16)(co[dt][r] / rsum[r]);
    }
}

// ---------------- launcher ----------------
extern "C" void kernel_launch(void* const* d_in, const int* in_sizes, int n_in,
                              void* d_out, int out_size, void* d_ws, size_t ws_size,
                              hipStream_t stream) {
  const float* x  = (const float*)d_in[0];   // [B,T,C]
  const float* Wa = (const float*)d_in[1];   // [C, 3C]
  const float* Wp = (const float*)d_in[2];   // [C, C]
  float* out = (float*)d_out;                // [B,T,C]

  char* wsp = (char*)d_ws;
  size_t off = 0;
  auto alloc = [&](size_t bytes) -> void* {
    void* p = wsp + off;
    off += (bytes + 255) & ~(size_t)255;
    return p;
  };
  bf16* xh   = (bf16*)alloc((size_t)BT_ * C_ * 2);        // 16 MB
  bf16* WaT  = (bf16*)alloc((size_t)3 * C_ * C_ * 2);     //  6 MB
  bf16* WpT  = (bf16*)alloc((size_t)C_ * C_ * 2);         //  2 MB
  bf16* Qh   = (bf16*)alloc((size_t)BT_ * C_ * 2);        // 16 MB
  bf16* Kh   = (bf16*)alloc((size_t)BT_ * C_ * 2);        // 16 MB
  bf16* Vt   = (bf16*)alloc((size_t)BT_ * C_ * 2);        // 16 MB
  bf16* Oh   = (bf16*)alloc((size_t)BT_ * C_ * 2);        // 16 MB
  float* qkv = (float*)alloc((size_t)BT_ * 3 * C_ * 4);   // 96 MB

  // 1. convert inputs to bf16 (weights transposed to [N,K])
  int n1 = BT_ * C_;
  f32_to_bf16_kernel<<<(n1 + 255) / 256, 256, 0, stream>>>(x, xh, n1);
  int n2 = 3 * C_ * C_;
  transpose_to_bf16_kernel<<<(n2 + 255) / 256, 256, 0, stream>>>(Wa, WaT, C_, 3 * C_);
  int n3 = C_ * C_;
  transpose_to_bf16_kernel<<<(n3 + 255) / 256, 256, 0, stream>>>(Wp, WpT, C_, C_);

  // 2. QKV GEMM: [8192,1024] @ [1024,3072] -> qkv f32
  gemm_bf16_kernel<<<dim3(3 * C_ / 128, BT_ / 128), 256, 0, stream>>>(
      xh, WaT, qkv, BT_, 3 * C_, C_);

  // 3. RoPE + head split
  rope_split_kernel<<<(n1 + 255) / 256, 256, 0, stream>>>(qkv, Qh, Kh, Vt);

  // 4. causal flash attention (8192 query tiles, 8 waves/block)
  flash_attn_kernel<<<(B_ * H_ * (T_ / 16)) / 8, 256, 0, stream>>>(Qh, Kh, Vt, Oh);

  // 5. projection GEMM: [8192,1024] @ [1024,1024] -> out f32
  gemm_bf16_kernel<<<dim3(C_ / 128, BT_ / 128), 256, 0, stream>>>(
      Oh, WpT, out, BT_, C_, C_);
}
